// LlamaAttention_30915174597032
// MI455X (gfx1250) — matrix operand order repaired and checked by reading, not compiled, not run on hardware
//
#include <hip/hip_runtime.h>

typedef __attribute__((ext_vector_type(16))) __bf16 v16bf;
typedef __attribute__((ext_vector_type(8)))  __bf16 v8bf;
typedef __attribute__((ext_vector_type(8)))  float  v8f;

constexpr int BATCH = 2, SEQ = 2048, DIM = 2048, NHEAD = 16, HDIM = 128;
constexpr int MROWS = BATCH * SEQ;              // 4096
constexpr float QSCALE = 0.08838834764831845f;  // 1/sqrt(128)

#define WMMA_BF16(a, b, c) \
  __builtin_amdgcn_wmma_f32_16x16x32_bf16(false, (a), false, (b), (short)0, (c), false, false)

// sched_group_barrier masks
#define SG_WMMA   0x008
#define SG_VMREAD 0x020
#define SG_DSREAD 0x100
#define SG_DSWRITE 0x200
#define SGB(mask, n, id) __builtin_amdgcn_sched_group_barrier((mask), (n), (id))

// Load one 16-element bf16 fragment as two 16-byte chunks (global or LDS).
__device__ __forceinline__ v16bf ld16(const __bf16* p0, const __bf16* p1) {
  union { v16bf v; v8bf h[2]; } u;
  u.h[0] = *(const v8bf*)p0;
  u.h[1] = *(const v8bf*)p1;
  return u.v;
}

// ---------------------------------------------------------------------------
// fp32 -> bf16 elementwise convert
__global__ void __launch_bounds__(256) k_f32_to_bf16(const float* __restrict__ in,
                                                     __bf16* __restrict__ out, int n) {
  int i = blockIdx.x * blockDim.x + threadIdx.x;
  if (i < n) out[i] = (__bf16)in[i];
}

// fp32 W[k][n] (DIM x DIM) -> bf16 Wt[n][k]
__global__ void __launch_bounds__(256) k_transpose_bf16(const float* __restrict__ in,
                                                        __bf16* __restrict__ out) {
  int i = blockIdx.x * blockDim.x + threadIdx.x;  // over DIM*DIM
  int k = i >> 11;
  int n = i & (DIM - 1);
  out[n * DIM + k] = (__bf16)in[i];
}

// ---------------------------------------------------------------------------
// C[4096,2048] = A[4096,2048]bf16 @ Bt[2048,2048]bf16 (Bt = B transposed).
// Block = 8 waves covering 128(M) x 128(N). 128x32 B panels in ping-pong LDS,
// branchless prefetch of chunk i+1 overlapping compute of chunk i, one
// barrier per chunk. sched_group_barrier pins: 16 DS reads -> 8 back-to-back
// WMMAs; the staging copy is an independent pipeline (syncid 1).
// mode: 0 = bf16 out (scaled), 1 = bf16 transposed per-head into
// Vt[b][h][d][s], 2 = f32 out.
__global__ void __launch_bounds__(256) k_gemm_bf16_wmma(const __bf16* __restrict__ A,
                                                        const __bf16* __restrict__ Bt,
                                                        void* __restrict__ Cout,
                                                        int mode, float scale) {
  __shared__ __align__(16) __bf16 Bs[2][128 * 32];  // ping-pong, 16 KB

  const int tid = threadIdx.x;
  const int wib = tid >> 5;
  const int lane = tid & 31;
  const int hf = lane >> 4;
  const int l16 = lane & 15;

  const int mb = blockIdx.x % (MROWS / 128);  // 32 M-blocks of 128 rows
  const int nb = blockIdx.x / (MROWS / 128);  // 16 N-blocks of 128 cols
  const int mr = mb * 128 + wib * 16;         // this wave's 16 rows
  const int nr = nb * 128;

  v8f acc[8] = {};

  const __bf16* arow = A + (size_t)(mr + l16) * DIM;
  // Cooperative B panel copy: thread t handles Bt[nr + t/2][(t&1)*16 .. +15]
  const int brow = tid >> 1;
  const int bkof = (tid & 1) * 16;
  const __bf16* bsrc = Bt + (size_t)(nr + brow) * DIM + bkof;
  const int bdof = brow * 32 + bkof;

  constexpr int NCH = DIM / 32;  // 64 K-chunks
  // prologue: stage chunk 0
  *(v8bf*)(Bs[0] + bdof) = *(const v8bf*)bsrc;
  *(v8bf*)(Bs[0] + bdof + 8) = *(const v8bf*)(bsrc + 8);

  for (int ic = 0; ic < NCH; ++ic) {
    __syncthreads();  // chunk ic staged; buffer (ic+1)&1 free
    // branchless prefetch (last iteration redundantly re-copies chunk 0)
    {
      const int icn = (ic + 1 < NCH) ? (ic + 1) : 0;
      const __bf16* s = bsrc + (size_t)icn * 32;
      __bf16* d = Bs[(ic + 1) & 1] + bdof;
      *(v8bf*)d = *(const v8bf*)s;
      *(v8bf*)(d + 8) = *(const v8bf*)(s + 8);
    }
    const __bf16* bbase = Bs[ic & 1];
    v16bf af = ld16(arow + ic * 32 + hf * 8, arow + ic * 32 + 16 + hf * 8);
    v16bf bfr[8];
#pragma unroll
    for (int t = 0; t < 8; ++t) {
      const __bf16* bp = bbase + (t * 16 + l16) * 32 + hf * 8;
      bfr[t] = ld16(bp, bp + 16);
    }
#pragma unroll
    for (int t = 0; t < 8; ++t) acc[t] = WMMA_BF16(af, bfr[t], acc[t]);

    // compute pipeline: A reads -> all 16 fragment reads -> 8 WMMAs
    SGB(SG_VMREAD, 2, 0);
    SGB(SG_DSREAD, 16, 0);
    SGB(SG_WMMA, 8, 0);
    // staging pipeline: panel reads then stores (drift late, overlap compute)
    SGB(SG_VMREAD, 2, 1);
    SGB(SG_DSWRITE, 2, 1);
  }

  if (mode == 0) {
    __bf16* C = (__bf16*)Cout;
#pragma unroll
    for (int t = 0; t < 8; ++t)
#pragma unroll
      for (int j = 0; j < 8; ++j)
        C[(size_t)(mr + j + 8 * hf) * DIM + nr + t * 16 + l16] = (__bf16)(acc[t][j] * scale);
  } else if (mode == 1) {  // scatter into Vt[b][h][d][s]
    __bf16* C = (__bf16*)Cout;
#pragma unroll
    for (int t = 0; t < 8; ++t)
#pragma unroll
      for (int j = 0; j < 8; ++j) {
        int m = mr + j + 8 * hf;
        int b = m >> 11, s = m & (SEQ - 1);
        int n = nr + t * 16 + l16;
        int h = n >> 7, d = n & (HDIM - 1);
        C[(size_t)((b * NHEAD + h) * HDIM + d) * SEQ + s] = (__bf16)acc[t][j];
      }
  } else {
    float* C = (float*)Cout;
#pragma unroll
    for (int t = 0; t < 8; ++t)
#pragma unroll
      for (int j = 0; j < 8; ++j)
        C[(size_t)(mr + j + 8 * hf) * DIM + nr + t * 16 + l16] = acc[t][j];
  }
}

// ---------------------------------------------------------------------------
// Flash attention. Block = 8 waves on the SAME (b, h), covering 128 queries.
// K (32x128) and transposed V (128x32) chunks staged in ping-pong LDS, one
// barrier per 32-key chunk, branchless prefetch. sched_group_barrier pins the
// two DS-read->WMMA chains; staging copy is an independent pipeline.
__global__ void __launch_bounds__(256) k_attn_wmma(const __bf16* __restrict__ Q,
                                                   const __bf16* __restrict__ Kmat,
                                                   const __bf16* __restrict__ Vt,
                                                   __bf16* __restrict__ Ctx) {
  __shared__ __align__(16) __bf16 Ks[2][32 * 128];  // [key][dim]  16 KB
  __shared__ __align__(16) __bf16 Vs[2][128 * 32];  // [dim][key]  16 KB
  __shared__ __align__(16) __bf16 Ps[8][16 * 32];   // per-wave P   8 KB

  const int tid = threadIdx.x;
  const int wib = tid >> 5;
  const int lane = tid & 31;
  const int hf = lane >> 4;
  const int l16 = lane & 15;

  const int qb = blockIdx.x % (SEQ / 128);  // 16 query-blocks of 128
  const int h = (blockIdx.x / (SEQ / 128)) % NHEAD;
  const int b = blockIdx.x / ((SEQ / 128) * NHEAD);
  const int qt = qb * 8 + wib;              // this wave's 16-query tile

  // Q fragments (head_dim 128 = 4 chunks of K=32), kept resident
  const __bf16* qrow = Q + (size_t)(b * SEQ + qt * 16 + l16) * DIM + h * HDIM;
  v16bf qf[4];
#pragma unroll
  for (int c = 0; c < 4; ++c)
    qf[c] = ld16(qrow + c * 32 + hf * 8, qrow + c * 32 + 16 + hf * 8);

  // Cooperative copy pointers.
  // K: thread t copies Kmat[key kc + t/8][dims (t&7)*16 .. +15]
  const int krow = tid >> 3;
  const int kof = (tid & 7) * 16;
  const __bf16* ksrc = Kmat + (size_t)(b * SEQ + krow) * DIM + h * HDIM + kof;
  const int kdof = krow * 128 + kof;
  // V: thread t copies Vt[dim t/2][keys kc + (t&1)*16 .. +15]
  const int vrow = tid >> 1;
  const int vof = (tid & 1) * 16;
  const __bf16* vsrc = Vt + (size_t)((b * NHEAD + h) * HDIM + vrow) * SEQ + vof;
  const int vdof = vrow * 32 + vof;

  v8f o[8] = {};
  float mrun[8], lrun[8];
#pragma unroll
  for (int j = 0; j < 8; ++j) { mrun[j] = -1e30f; lrun[j] = 0.0f; }

  constexpr int NCH = SEQ / 32;  // 64 key chunks
  // prologue: stage chunk 0
  *(v8bf*)(Ks[0] + kdof) = *(const v8bf*)ksrc;
  *(v8bf*)(Ks[0] + kdof + 8) = *(const v8bf*)(ksrc + 8);
  *(v8bf*)(Vs[0] + vdof) = *(const v8bf*)vsrc;
  *(v8bf*)(Vs[0] + vdof + 8) = *(const v8bf*)(vsrc + 8);

  for (int ic = 0; ic < NCH; ++ic) {
    __syncthreads();  // chunk ic staged; other buffer free
    // branchless prefetch (last iteration redundantly re-copies chunk 0)
    {
      const int icn = (ic + 1 < NCH) ? (ic + 1) : 0;
      const int nbuf = (ic + 1) & 1;
      const __bf16* s0p = ksrc + (size_t)icn * 32 * DIM;
      __bf16* d0 = Ks[nbuf] + kdof;
      *(v8bf*)d0 = *(const v8bf*)s0p;
      *(v8bf*)(d0 + 8) = *(const v8bf*)(s0p + 8);
      const __bf16* s1p = vsrc + icn * 32;
      __bf16* d1 = Vs[nbuf] + vdof;
      *(v8bf*)d1 = *(const v8bf*)s1p;
      *(v8bf*)(d1 + 8) = *(const v8bf*)(s1p + 8);
    }
    const __bf16* kbase = Ks[ic & 1];
    const __bf16* vbase = Vs[ic & 1];

    // scores for keys [kc, kc+32): two 16x16 tiles; preload all 8 K-fragments
    v16bf kfr[8];
#pragma unroll
    for (int c = 0; c < 4; ++c) {
      const __bf16* kp0 = kbase + l16 * 128 + c * 32 + hf * 8;
      kfr[2 * c] = ld16(kp0, kp0 + 16);
      const __bf16* kp1 = kbase + (16 + l16) * 128 + c * 32 + hf * 8;
      kfr[2 * c + 1] = ld16(kp1, kp1 + 16);
    }
    v8f s0 = {}, s1 = {};
#pragma unroll
    for (int c = 0; c < 4; ++c) {
      s0 = WMMA_BF16(qf[c], kfr[2 * c], s0);
      s1 = WMMA_BF16(qf[c], kfr[2 * c + 1], s1);
    }

    // online softmax update (row j lives on lanes sharing this half-wave)
    float alpha[8];
#pragma unroll
    for (int j = 0; j < 8; ++j) {
      float cm = fmaxf(s0[j], s1[j]);
#pragma unroll
      for (int mk = 1; mk <= 8; mk <<= 1) cm = fmaxf(cm, __shfl_xor(cm, mk, 32));
      float mnew = fmaxf(mrun[j], cm);
      alpha[j] = __expf(mrun[j] - mnew);
      float p0 = __expf(s0[j] - mnew);
      float p1 = __expf(s1[j] - mnew);
      s0[j] = p0; s1[j] = p1;
      float rs = p0 + p1;
#pragma unroll
      for (int mk = 1; mk <= 8; mk <<= 1) rs += __shfl_xor(rs, mk, 32);
      lrun[j] = lrun[j] * alpha[j] + rs;
      mrun[j] = mnew;
    }
#pragma unroll
    for (int t = 0; t < 8; ++t)
#pragma unroll
      for (int j = 0; j < 8; ++j) o[t][j] *= alpha[j];

    // re-stripe P (C-layout) -> A-fragment layout via per-wave LDS region;
    // same-wave LDS ops are in-order (DScnt), no block barrier needed.
    __bf16* pl = Ps[wib];
#pragma unroll
    for (int j = 0; j < 8; ++j) {
      pl[(j + 8 * hf) * 32 + l16] = (__bf16)s0[j];
      pl[(j + 8 * hf) * 32 + 16 + l16] = (__bf16)s1[j];
    }
    v16bf pf = ld16(pl + l16 * 32 + hf * 8, pl + l16 * 32 + 16 + hf * 8);

    // O += P @ V  (8 output tiles cover head_dim 128); preload V fragments
    v16bf vfr[8];
#pragma unroll
    for (int t = 0; t < 8; ++t) {
      const __bf16* vp = vbase + (t * 16 + l16) * 32 + hf * 8;
      vfr[t] = ld16(vp, vp + 16);
    }
#pragma unroll
    for (int t = 0; t < 8; ++t) o[t] = WMMA_BF16(pf, vfr[t], o[t]);

    // compute pipeline: K reads -> QK WMMAs -> P stores -> P+V reads -> PV WMMAs
    SGB(SG_DSREAD, 16, 0);
    SGB(SG_WMMA, 8, 0);
    SGB(SG_DSWRITE, 16, 0);
    SGB(SG_DSREAD, 18, 0);
    SGB(SG_WMMA, 8, 0);
    // staging pipeline: global reads then LDS stores (drift late)
    SGB(SG_VMREAD, 4, 1);
    SGB(SG_DSWRITE, 4, 1);
  }

#pragma unroll
  for (int j = 0; j < 8; ++j) lrun[j] = 1.0f / lrun[j];
  const int rowbase = b * SEQ + qt * 16;
#pragma unroll
  for (int t = 0; t < 8; ++t)
#pragma unroll
    for (int j = 0; j < 8; ++j)
      Ctx[(size_t)(rowbase + j + 8 * hf) * DIM + h * HDIM + t * 16 + l16] =
          (__bf16)(o[t][j] * lrun[j]);
}

// ---------------------------------------------------------------------------
extern "C" void kernel_launch(void* const* d_in, const int* in_sizes, int n_in,
                              void* d_out, int out_size, void* d_ws, size_t ws_size,
                              hipStream_t stream) {
  (void)in_sizes; (void)n_in; (void)out_size; (void)ws_size;
  const float* x  = (const float*)d_in[0];
  const float* Wq = (const float*)d_in[1];
  const float* Wk = (const float*)d_in[2];
  const float* Wv = (const float*)d_in[3];
  const float* Wo = (const float*)d_in[4];
  float* out = (float*)d_out;

  char* ws = (char*)d_ws;
  const size_t MB = 1u << 20;
  __bf16* xb  = (__bf16*)(ws + 0 * MB);    // 16 MB  [4096,2048]
  __bf16* WqT = (__bf16*)(ws + 16 * MB);   //  8 MB  [2048,2048] transposed
  __bf16* WkT = (__bf16*)(ws + 24 * MB);
  __bf16* WvT = (__bf16*)(ws + 32 * MB);
  __bf16* WoT = (__bf16*)(ws + 40 * MB);
  __bf16* Qb  = (__bf16*)(ws + 48 * MB);   // 16 MB (pre-scaled by 1/sqrt(d))
  __bf16* Kb  = (__bf16*)(ws + 64 * MB);   // 16 MB
  __bf16* Vt  = (__bf16*)(ws + 80 * MB);   // 16 MB [B][H][HDIM][SEQ]
  __bf16* Ctx = (__bf16*)(ws + 96 * MB);   // 16 MB

  const int nX = MROWS * DIM;          // 8,388,608
  const int nW = DIM * DIM;            // 4,194,304
  k_f32_to_bf16<<<(nX + 255) / 256, 256, 0, stream>>>(x, xb, nX);
  k_transpose_bf16<<<nW / 256, 256, 0, stream>>>(Wq, WqT);
  k_transpose_bf16<<<nW / 256, 256, 0, stream>>>(Wk, WkT);
  k_transpose_bf16<<<nW / 256, 256, 0, stream>>>(Wv, WvT);
  k_transpose_bf16<<<nW / 256, 256, 0, stream>>>(Wo, WoT);

  // 512 blocks x 256 threads: 32 M-blocks x 16 N-blocks
  k_gemm_bf16_wmma<<<512, 256, 0, stream>>>(xb, WqT, Qb, 0, QSCALE);
  k_gemm_bf16_wmma<<<512, 256, 0, stream>>>(xb, WkT, Kb, 0, 1.0f);
  k_gemm_bf16_wmma<<<512, 256, 0, stream>>>(xb, WvT, Vt, 1, 1.0f);

  // 512 blocks: (b, h, 128-query block); 8 waves share staged K/V chunks
  k_attn_wmma<<<512, 256, 0, stream>>>(Qb, Kb, Vt, Ctx);

  k_gemm_bf16_wmma<<<512, 256, 0, stream>>>(Ctx, WoT, out, 2, 1.0f);
}
